// encoder_66546223284401
// MI455X (gfx1250) — compile-verified, matrix operand order checked by reading, not run on hardware
//
#include <hip/hip_runtime.h>

typedef unsigned short ushortT;
typedef __bf16 v16bf __attribute__((ext_vector_type(16)));
typedef float  v8f   __attribute__((ext_vector_type(8)));
typedef unsigned int v8u __attribute__((ext_vector_type(8)));

#define B_   32
#define T_   1024
#define D_   512
#define HL_  512
#define HH_  1024
#define G4L  (4*HL_)   // 2048
#define G4H  (4*HH_)   // 4096
#define NCH  16
#define RNWG 16        // workgroups per direction in the recurrent kernel

// ---------- small helpers ----------
__device__ __forceinline__ float bf2f(ushortT u) {
  unsigned v = ((unsigned)u) << 16;
  return __builtin_bit_cast(float, v);
}
__device__ __forceinline__ ushortT f2bf(float f) {
  unsigned u = __builtin_bit_cast(unsigned, f);
  u += 0x7FFFu + ((u >> 16) & 1u);           // round-to-nearest-even
  return (ushortT)(u >> 16);
}
__device__ __forceinline__ float sigmoidf_(float x) { return 1.0f / (1.0f + __expf(-x)); }
// K-pair byte layout for 16-bit WMMA operands (ISA 7.12.2): VGPR v holds K pair
__device__ __forceinline__ int kpair(int v, int hl) {
  return (v < 4 ? 2 * v : 16 + 2 * (v - 4)) + hl * 8;
}
__device__ __forceinline__ v8f wmma_bf16(v16bf a, v16bf b, v8f c) {
  return __builtin_amdgcn_wmma_f32_16x16x32_bf16(false, a, false, b, (short)0, c, false, false);
}

// CDNA5 async global->LDS DMA (16B per lane, tracked by ASYNCcnt)
__device__ __forceinline__ void async_ld_b128(const void* lds_dst, const void* gsrc) {
  unsigned lds_off = (unsigned)(size_t)lds_dst;             // low 32 bits = LDS byte offset
  unsigned long long ga = (unsigned long long)(size_t)gsrc;
  asm volatile("global_load_async_to_lds_b128 %0, %1, off"
               :: "v"(lds_off), "v"(ga) : "memory");
}
__device__ __forceinline__ void wait_async0() {
  asm volatile("s_wait_asynccnt 0x0" ::: "memory");
}

// ---------- f32 -> bf16 conversion (vectorized: 8 elems / thread) ----------
__global__ void conv_bf16_kernel(const float* __restrict__ src,
                                 ushortT* __restrict__ dst, size_t n) {
  size_t i = ((size_t)blockIdx.x * blockDim.x + threadIdx.x) * 8;
  if (i + 8 <= n) {
    float4 a = *(const float4*)(src + i);
    float4 b = *(const float4*)(src + i + 4);
    ushortT o[8] = {f2bf(a.x), f2bf(a.y), f2bf(a.z), f2bf(a.w),
                    f2bf(b.x), f2bf(b.y), f2bf(b.z), f2bf(b.w)};
    *(uint4*)(dst + i) = *(const uint4*)o;
  } else {
    for (; i < n; ++i) dst[i] = f2bf(src[i]);
  }
}

// ---------- generic WMMA GEMM: C[M,N] = A[M,K](bf16) * W[N,K]^T(bf16) (+bias0+bias1) (+=) ----------
// block tile 32x256, 8 waves; software-pipelined global->reg->LDS staging + prefetch
__global__ __launch_bounds__(256) void gemm_bf16_kernel(
    const ushortT* __restrict__ A, int lda,
    const ushortT* __restrict__ W, int ldw,
    float* __restrict__ outF, ushortT* __restrict__ outB,
    const float* __restrict__ bias0, const float* __restrict__ bias1,
    int M, int N, int K, int accumulate)
{
  __shared__ ushortT Ash[32 * 40];    // 32x32 tile, stride 40 (bank-conflict pad)
  __shared__ ushortT Wsh[256 * 40];   // 256x32 tile

  const int tid  = threadIdx.x;
  const int wv   = tid >> 5;
  const int lane = tid & 31;
  const int hl   = lane >> 4;
  const int lm   = lane & 15;
  const int M0 = blockIdx.y * 32;
  const int N0 = blockIdx.x * 256;
  const int ar = tid >> 2, aseg = tid & 3;   // A-tile loader role (tid < 128)

  v8f acc[2][2];
#pragma unroll
  for (int i = 0; i < 2; i++)
#pragma unroll
    for (int j = 0; j < 2; j++)
#pragma unroll
      for (int v = 0; v < 8; v++) acc[i][j][v] = 0.0f;

  uint4 aReg = {0, 0, 0, 0};
  uint4 wReg[4];
  // prologue: fetch first K tile into registers
  if (tid < 128)
    aReg = *(const uint4*)(A + (size_t)(M0 + ar) * lda + aseg * 8);
#pragma unroll
  for (int j = 0; j < 4; j++) {
    int idx = j * 256 + tid;
    int r = idx >> 2, seg = idx & 3;
    wReg[j] = *(const uint4*)(W + (size_t)(N0 + r) * ldw + seg * 8);
  }

  for (int k0 = 0; k0 < K; k0 += 32) {
    __syncthreads();                         // prior fragment reads done
    if (tid < 128) *(uint4*)(Ash + ar * 40 + aseg * 8) = aReg;
#pragma unroll
    for (int j = 0; j < 4; j++) {
      int idx = j * 256 + tid;
      int r = idx >> 2, seg = idx & 3;
      *(uint4*)(Wsh + r * 40 + seg * 8) = wReg[j];
    }
    __syncthreads();                         // tiles visible

    const int kn = k0 + 32;
    if (kn < K) {                            // overlap next-tile fetch with WMMA
      if (tid < 128)
        aReg = *(const uint4*)(A + (size_t)(M0 + ar) * lda + kn + aseg * 8);
#pragma unroll
      for (int j = 0; j < 4; j++) {
        int idx = j * 256 + tid;
        int r = idx >> 2, seg = idx & 3;
        wReg[j] = *(const uint4*)(W + (size_t)(N0 + r) * ldw + kn + seg * 8);
      }
      if (kn + 32 < K)                       // prime GL2 for the tile after next
        __builtin_prefetch(W + (size_t)(N0 + tid) * ldw + kn + 32, 0, 3);
    }

    v16bf aF[2], bF[2];
#pragma unroll
    for (int mt = 0; mt < 2; mt++) {
      v8u u;
#pragma unroll
      for (int v = 0; v < 8; v++)
        u[v] = *(const unsigned*)(Ash + (mt * 16 + lm) * 40 + kpair(v, hl));
      aF[mt] = __builtin_bit_cast(v16bf, u);
    }
#pragma unroll
    for (int nt = 0; nt < 2; nt++) {
      v8u u;
      int row = wv * 32 + nt * 16 + lm;
#pragma unroll
      for (int v = 0; v < 8; v++)
        u[v] = *(const unsigned*)(Wsh + row * 40 + kpair(v, hl));
      bF[nt] = __builtin_bit_cast(v16bf, u);
    }
#pragma unroll
    for (int mt = 0; mt < 2; mt++)
#pragma unroll
      for (int nt = 0; nt < 2; nt++)
        acc[mt][nt] = wmma_bf16(aF[mt], bF[nt], acc[mt][nt]);
  }

#pragma unroll
  for (int mt = 0; mt < 2; mt++) {
#pragma unroll
    for (int nt = 0; nt < 2; nt++) {
      int col = N0 + wv * 32 + nt * 16 + lm;
      float b0 = bias0 ? bias0[col] : 0.0f;
      float b1 = bias1 ? bias1[col] : 0.0f;
#pragma unroll
      for (int v = 0; v < 8; v++) {
        int row = M0 + mt * 16 + v + hl * 8;
        if (row < M) {
          float val = acc[mt][nt][v] + b0 + b1;
          size_t idx = (size_t)row * N + col;
          if (outF) {
            if (accumulate) val += outF[idx];
            outF[idx] = val;
          }
          if (outB) outB[idx] = f2bf(val);
        }
      }
    }
  }
}

// ---------- persistent low-level bi-LSTM scan ----------
// grid (16, 2): blockIdx.y = direction, 16 WGs each own 32 hidden units (all 4 gates).
// Whh slice (128KB) staged ONCE in LDS via async DMA (CDNA5 WGP has 320KB);
// h broadcast via L2-resident global buffer, async-DMA'd to LDS each step;
// 2-phase grid barrier per step; c-state in registers.
__global__ __launch_bounds__(256) void lstm_low_kernel(
    const ushortT* __restrict__ xgF, const ushortT* __restrict__ xgB,
    const ushortT* __restrict__ WhhF, const ushortT* __restrict__ WhhB,
    ushortT* __restrict__ h_buf,
    float* __restrict__ out_low, ushortT* __restrict__ outlow_bf,
    unsigned* __restrict__ bar)
{
  extern __shared__ char lstm_smem[];
  ushortT* wsh = (ushortT*)lstm_smem;                         // 128 x 512, stride 520
  ushortT* hsh = (ushortT*)(lstm_smem + 128 * 520 * 2);       // 32 x 512, stride 520
  float*   gsh = (float*)(lstm_smem + (128 + 32) * 520 * 2);  // 32 x 128, stride 132

  const int dir   = blockIdx.y;
  const int wg    = blockIdx.x;
  const int hbase = wg * 32;
  const ushortT* xg  = dir ? xgB : xgF;
  const ushortT* Whh = dir ? WhhB : WhhF;
  ushortT* hb = h_buf + (size_t)dir * (B_ * HL_);
  volatile unsigned* mybar = bar + (size_t)dir * (2 * T_);

  const int tid  = threadIdx.x;
  const int wv   = tid >> 5;
  const int lane = tid & 31;
  const int hl   = lane >> 4;
  const int lm   = lane & 15;
  const int lr   = wv * 16 + lm;      // this lane's local gate column (0..127)

  // one-time stage of this WG's Whh slice into LDS via async global->LDS DMA
#pragma unroll 4
  for (int rep = 0; rep < 32; rep++) {
    int idx = rep * 256 + tid;        // 8192 x B128 transfers
    int r2 = idx >> 6, s4 = idx & 63;
    int qq = r2 >> 5, j2 = r2 & 31;
    int gr = qq * HL_ + hbase + j2;
    async_ld_b128(wsh + r2 * 520 + s4 * 8, Whh + (size_t)gr * HL_ + s4 * 8);
  }
  wait_async0();

  float creg[4] = {0.f, 0.f, 0.f, 0.f};   // persistent cell state

  for (int t = 0; t < T_; t++) {
    const int trow = dir ? (T_ - 1 - t) : t;

    __syncthreads();
#pragma unroll
    for (int rep = 0; rep < 8; rep++) {            // async-DMA h -> LDS
      int idx = rep * 256 + tid;
      int r = idx >> 6, s4 = idx & 63;
      async_ld_b128(hsh + r * 520 + s4 * 8, hb + r * 512 + s4 * 8);
    }
    wait_async0();
    __syncthreads();

    // grid barrier phase A: everyone has consumed h(t-1)
    {
      __threadfence();
      __syncthreads();
      if (tid == 0) {
        atomicAdd((unsigned*)(mybar + 2 * t), 1u);
        while (mybar[2 * t] < RNWG) __builtin_amdgcn_s_sleep(1);
      }
      __syncthreads();
    }

    v8f acc[2];
#pragma unroll
    for (int i = 0; i < 2; i++)
#pragma unroll
      for (int v = 0; v < 8; v++) acc[i][v] = 0.0f;

    for (int kt = 0; kt < 16; kt++) {
      v16bf bF;
      {
        v8u u;
#pragma unroll
        for (int v = 0; v < 8; v++)
          u[v] = *(const unsigned*)(wsh + lr * 520 + kt * 32 + kpair(v, hl));
        bF = __builtin_bit_cast(v16bf, u);
      }
#pragma unroll
      for (int mt = 0; mt < 2; mt++) {
        v8u u;
#pragma unroll
        for (int v = 0; v < 8; v++)
          u[v] = *(const unsigned*)(hsh + (mt * 16 + lm) * 520 + kt * 32 + kpair(v, hl));
        v16bf aF = __builtin_bit_cast(v16bf, u);
        acc[mt] = wmma_bf16(aF, bF, acc[mt]);
      }
    }

#pragma unroll
    for (int mt = 0; mt < 2; mt++)
#pragma unroll
      for (int v = 0; v < 8; v++)
        gsh[(mt * 16 + v + hl * 8) * 132 + wv * 16 + lm] = acc[mt][v];
    __syncthreads();

#pragma unroll
    for (int e = 0; e < 4; e++) {                  // cell update, 4 (b,j) per thread
      int idx = tid * 4 + e;
      int b = idx >> 5, j = idx & 31;
      size_t xbase = ((size_t)b * T_ + trow) * G4L + hbase + j;
      float gi = gsh[b * 132 +       j] + bf2f(xg[xbase]);
      float gf = gsh[b * 132 +  32 + j] + bf2f(xg[xbase + 512]);
      float gc = gsh[b * 132 +  64 + j] + bf2f(xg[xbase + 1024]);
      float go = gsh[b * 132 +  96 + j] + bf2f(xg[xbase + 1536]);
      float c = sigmoidf_(gf) * creg[e] + sigmoidf_(gi) * tanhf(gc);
      creg[e] = c;
      float h = sigmoidf_(go) * tanhf(c);
      hb[b * HL_ + hbase + j] = f2bf(h);
      size_t oidx = ((size_t)b * T_ + trow) * (2 * HL_) + dir * HL_ + hbase + j;
      out_low[oidx]   = h;
      outlow_bf[oidx] = f2bf(h);
    }

    // grid barrier phase B: everyone has published h(t)
    {
      __threadfence();
      __syncthreads();
      if (tid == 0) {
        atomicAdd((unsigned*)(mybar + 2 * t + 1), 1u);
        while (mybar[2 * t + 1] < RNWG) __builtin_amdgcn_s_sleep(1);
      }
      __syncthreads();
    }
  }
}

// ---------- attention (per chunk): scores -> softmax -> context mix ----------
__global__ __launch_bounds__(256) void attn_kernel(
    const float* __restrict__ v1, const float* __restrict__ v2,
    const float* __restrict__ w_attn,
    const ushortT* __restrict__ outlow_bf, ushortT* __restrict__ xin_bf, int chunk)
{
  __shared__ float part[256];
  __shared__ float attn[64];
  const int b = blockIdx.x, tid = threadIdx.x;
  const int base = chunk * 64;
  const int t = tid >> 2, p = tid & 3;
  const float* v1r = v1 + (size_t)b * HL_;
  const float* v2r = v2 + ((size_t)b * T_ + base + t) * HL_;
  float s = 0.f;
  for (int h = p; h < HL_; h += 4) s += tanhf(v1r[h] + v2r[h]) * w_attn[h];
  part[tid] = s;
  __syncthreads();
  if (tid < 64) attn[tid] = part[tid*4] + part[tid*4+1] + part[tid*4+2] + part[tid*4+3];
  __syncthreads();
  if (tid == 0) {
    float m = attn[0];
    for (int i = 1; i < 64; i++) m = fmaxf(m, attn[i]);
    float sum = 0.f;
    for (int i = 0; i < 64; i++) { float e = __expf(attn[i] - m); attn[i] = e; sum += e; }
    float inv = 1.f / sum;
    for (int i = 0; i < 64; i++) attn[i] *= inv;
  }
  __syncthreads();
  for (int d = tid; d < 2 * HL_; d += 256) {
    float a = 0.f;
    for (int tt = 0; tt < 64; tt++)
      a += attn[tt] * bf2f(outlow_bf[((size_t)b * T_ + base + tt) * (2 * HL_) + d]);
    xin_bf[(size_t)b * (2 * HL_) + d] = f2bf(a);
  }
}

// ---------- high-level cell update (c0 == 0, so f-gate term vanishes) ----------
__global__ void cell_high_kernel(const float* __restrict__ g,
                                 float* __restrict__ out_high,
                                 ushortT* __restrict__ hh_bf, int chunk)
{
  int e = blockIdx.x * blockDim.x + threadIdx.x;
  if (e >= B_ * HH_) return;
  int b = e >> 10, u = e & (HH_ - 1);
  const float* gr = g + (size_t)b * G4H;
  float c = sigmoidf_(gr[u]) * tanhf(gr[2 * HH_ + u]);
  float h = sigmoidf_(gr[3 * HH_ + u]) * tanhf(c);
  out_high[((size_t)b * NCH + chunk) * HH_ + u] = h;
  hh_bf[(size_t)b * HH_ + u] = f2bf(h);
}

// ---------- host orchestration ----------
extern "C" void kernel_launch(void* const* d_in, const int* in_sizes, int n_in,
                              void* d_out, int out_size, void* d_ws, size_t ws_size,
                              hipStream_t stream) {
  (void)in_sizes; (void)n_in; (void)out_size; (void)ws_size;
  const float* x      = (const float*)d_in[0];
  const float* Wih_f  = (const float*)d_in[1];
  const float* Whh_f  = (const float*)d_in[2];
  const float* bih_f  = (const float*)d_in[3];
  const float* bhh_f  = (const float*)d_in[4];
  const float* Wih_b  = (const float*)d_in[5];
  const float* Whh_b  = (const float*)d_in[6];
  const float* bih_b  = (const float*)d_in[7];
  const float* bhh_b  = (const float*)d_in[8];
  const float* Wih_h  = (const float*)d_in[9];
  const float* Whh_h  = (const float*)d_in[10];
  const float* bih_h  = (const float*)d_in[11];
  const float* bhh_h  = (const float*)d_in[12];
  const float* Wla    = (const float*)d_in[13];
  const float* Wha    = (const float*)d_in[14];
  const float* w_attn = (const float*)d_in[15];

  char* cur = (char*)d_ws;
  auto alloc = [&](size_t bytes) -> char* {
    char* p = cur; cur += (bytes + 255) & ~(size_t)255; return p;
  };
  const size_t nBT = (size_t)B_ * T_;
  ushortT* x_bf    = (ushortT*)alloc(nBT * D_ * 2);
  ushortT* WihF_bf = (ushortT*)alloc((size_t)G4L * D_ * 2);
  ushortT* WihB_bf = (ushortT*)alloc((size_t)G4L * D_ * 2);
  ushortT* WhhF_bf = (ushortT*)alloc((size_t)G4L * HL_ * 2);
  ushortT* WhhB_bf = (ushortT*)alloc((size_t)G4L * HL_ * 2);
  ushortT* WihH_bf = (ushortT*)alloc((size_t)G4H * (2 * HL_) * 2);
  ushortT* WhhH_bf = (ushortT*)alloc((size_t)G4H * HH_ * 2);
  ushortT* Wla_bf  = (ushortT*)alloc((size_t)HL_ * (2 * HL_) * 2);
  ushortT* Wha_bf  = (ushortT*)alloc((size_t)HL_ * HH_ * 2);
  ushortT* xgF_bf  = (ushortT*)alloc(nBT * G4L * 2);
  ushortT* xgB_bf  = (ushortT*)alloc(nBT * G4L * 2);
  ushortT* outlow_bf = (ushortT*)alloc(nBT * (2 * HL_) * 2);
  float*   v2_buf  = (float*)alloc(nBT * HL_ * 4);
  ushortT* h_buf   = (ushortT*)alloc(2 * (size_t)B_ * HL_ * 2);
  ushortT* hh_bf   = (ushortT*)alloc((size_t)B_ * HH_ * 2);
  ushortT* xin_bf  = (ushortT*)alloc((size_t)B_ * (2 * HL_) * 2);
  float*   v1_buf  = (float*)alloc((size_t)B_ * HL_ * 4);
  float*   g_buf   = (float*)alloc((size_t)B_ * G4H * 4);
  unsigned* bar    = (unsigned*)alloc(2 * 2 * (size_t)T_ * 4);

  hipMemsetAsync(h_buf, 0, 2 * (size_t)B_ * HL_ * 2, stream);
  hipMemsetAsync(hh_bf, 0, (size_t)B_ * HH_ * 2, stream);
  hipMemsetAsync(bar, 0, 2 * 2 * (size_t)T_ * 4, stream);

  auto conv = [&](const float* s, ushortT* d, size_t n) {
    conv_bf16_kernel<<<(unsigned)((n / 8 + 255) / 256), 256, 0, stream>>>(s, d, n);
  };
  conv(x, x_bf, nBT * D_);
  conv(Wih_f, WihF_bf, (size_t)G4L * D_);
  conv(Wih_b, WihB_bf, (size_t)G4L * D_);
  conv(Whh_f, WhhF_bf, (size_t)G4L * HL_);
  conv(Whh_b, WhhB_bf, (size_t)G4L * HL_);
  conv(Wih_h, WihH_bf, (size_t)G4H * (2 * HL_));
  conv(Whh_h, WhhH_bf, (size_t)G4H * HH_);
  conv(Wla, Wla_bf, (size_t)HL_ * (2 * HL_));
  conv(Wha, Wha_bf, (size_t)HL_ * HH_);

  // input-gate GEMMs (bias folded; stored bf16 for the recurrent kernel)
  gemm_bf16_kernel<<<dim3(G4L / 256, (unsigned)(nBT / 32)), 256, 0, stream>>>(
      x_bf, D_, WihF_bf, D_, nullptr, xgF_bf, bih_f, bhh_f, (int)nBT, G4L, D_, 0);
  gemm_bf16_kernel<<<dim3(G4L / 256, (unsigned)(nBT / 32)), 256, 0, stream>>>(
      x_bf, D_, WihB_bf, D_, nullptr, xgB_bf, bih_b, bhh_b, (int)nBT, G4L, D_, 0);

  // persistent bidirectional recurrent scan (Whh staged in 320KB WGP LDS)
  const size_t lstm_lds = (size_t)(128 + 32) * 520 * 2 + (size_t)32 * 132 * 4;
  float* out_low = (float*)d_out;
  lstm_low_kernel<<<dim3(RNWG, 2), 256, lstm_lds, stream>>>(
      xgF_bf, xgB_bf, WhhF_bf, WhhB_bf, h_buf, out_low, outlow_bf, bar);

  // hoisted attention key projection: v2_all = outputs_low @ W_low_attn^T
  gemm_bf16_kernel<<<dim3(HL_ / 256, (unsigned)(nBT / 32)), 256, 0, stream>>>(
      outlow_bf, 2 * HL_, Wla_bf, 2 * HL_, v2_buf, nullptr, nullptr, nullptr,
      (int)nBT, HL_, 2 * HL_, 0);

  float* out_high = (float*)d_out + nBT * (2 * HL_);
  for (int n = 0; n < NCH; n++) {
    gemm_bf16_kernel<<<dim3(HL_ / 256, 1), 256, 0, stream>>>(
        hh_bf, HH_, Wha_bf, HH_, v1_buf, nullptr, nullptr, nullptr, B_, HL_, HH_, 0);
    attn_kernel<<<B_, 256, 0, stream>>>(v1_buf, v2_buf, w_attn, outlow_bf, xin_bf, n);
    gemm_bf16_kernel<<<dim3(G4H / 256, 1), 256, 0, stream>>>(
        xin_bf, 2 * HL_, WihH_bf, 2 * HL_, g_buf, nullptr, bih_h, bhh_h, B_, G4H, 2 * HL_, 0);
    gemm_bf16_kernel<<<dim3(G4H / 256, 1), 256, 0, stream>>>(
        hh_bf, HH_, WhhH_bf, HH_, g_buf, nullptr, nullptr, nullptr, B_, G4H, HH_, 1);
    cell_high_kernel<<<(B_ * HH_) / 256, 256, 0, stream>>>(g_buf, out_high, hh_bf, n);
  }
}
